// MPNNBlock_26010321944813
// MI455X (gfx1250) — matrix operand order repaired and checked by reading, not compiled, not run on hardware
//
#include <hip/hip_runtime.h>
#include <hip/hip_bf16.h>

typedef __bf16 bf16;
typedef __attribute__((ext_vector_type(16))) __bf16 v16bf;
typedef __attribute__((ext_vector_type(8)))  float  v8f;

#define NN      50000
#define NE      1600000
#define NDIM    128
#define EDIM    16
#define HID     64
#define LN_EPS  1e-5f
#define ETILE   64

#define WMMA_BF16(a, b, c) \
  __builtin_amdgcn_wmma_f32_16x16x32_bf16(false, (a), false, (b), (short)0, (c), false, false)

union V16BF { v16bf v; uint4 q[2]; };

// ---- operand fragment loaders (CDNA5 ISA 7.12.2, wave32) --------------------
// A (16x32 bf16): lane holds row M = lane&15; K = khalf + {0..7, 16..23}
// (khalf = 8*(lane>=16)).  The 16 elements are two contiguous 16B runs.

static __device__ inline v16bf load_A_b128(const bf16* base, int ld, int kbase, int lane) {
  const int row   = lane & 15;
  const int khalf = (lane >> 4) << 3;
  const bf16* p = base + row * ld + kbase + khalf;
  V16BF t;
  t.q[0] = *(const uint4*)p;
  t.q[1] = *(const uint4*)(p + 16);
  return t.v;
}

// B fragments pre-packed to [(kt*4+nt)*32 + lane][16] bf16 (zero-padded K):
// one lane's fragment is 32 contiguous bytes.
static __device__ inline v16bf load_B_packed(const bf16* __restrict__ base, int tile, int lane) {
  const uint4* p = (const uint4*)(base + (((tile << 5) + lane)) * 16);
  V16BF t;
  t.q[0] = p[0];
  t.q[1] = p[1];
  return t.v;
}

// C/D (16x16 f32): lane holds column N = lane&15, rows M = j + 8*(lane>=16).
static __device__ inline v8f bias_acc(const float* __restrict__ bias, int nbase, int lane) {
  const float bv = bias[nbase + (lane & 15)];
  v8f c;
#pragma unroll
  for (int j = 0; j < 8; ++j) c[j] = bv;
  return c;
}

// ---- weight pre-pack: f32 [K x 64] -> bf16 B-operand layout -----------------

__global__ void pack_B(const float* __restrict__ W, bf16* __restrict__ out,
                       int K, int ktiles) {
  const int idx = blockIdx.x * blockDim.x + threadIdx.x;
  const int total = ktiles * 4 * 512;
  if (idx >= total) return;
  const int i    = idx & 15;
  const int lane = (idx >> 4) & 31;
  const int tile = idx >> 9;
  const int nt   = tile & 3;
  const int kt   = tile >> 2;
  const int col  = nt * 16 + (lane & 15);
  const int k    = kt * 32 + ((lane >> 4) << 3) + (i & 7) + ((i >> 3) << 4);
  out[idx] = (bf16)((k < K) ? W[k * HID + col] : 0.0f);
}

// ---- small utility kernels --------------------------------------------------

__global__ void fill_f32(float* __restrict__ p, float v, int n) {
  const int i = blockIdx.x * blockDim.x + threadIdx.x;
  if (i < n) p[i] = v;
}

__global__ void f32_to_bf16_k(const float* __restrict__ in, bf16* __restrict__ out, int n) {
  const int i = blockIdx.x * blockDim.x + threadIdx.x;
  if (i < n) out[i] = (bf16)in[i];
}

__global__ void degree_k(const int* __restrict__ dst, float* __restrict__ cnt, int e) {
  const int i = blockIdx.x * blockDim.x + threadIdx.x;
  if (i < e) atomicAdd(&cnt[dst[i]], 1.0f);
}

__global__ void invert_k(float* __restrict__ c, int n) {
  const int i = blockIdx.x * blockDim.x + threadIdx.x;
  if (i < n) c[i] = 1.0f / c[i];   // cnt >= 1 (self loops)
}

// ---- xt = x @ Wn + bn  (bf16 output for downstream WMMA gathers) ------------

__global__ __launch_bounds__(128)
void node_xform(const bf16* __restrict__ xin, const bf16* __restrict__ pWn,
                const float* __restrict__ b, bf16* __restrict__ xt_bf,
                int n, int din) {
  const int lane = threadIdx.x & 31;
  const int wave = blockIdx.x * (blockDim.x >> 5) + (threadIdx.x >> 5);
  const int tb   = wave * 16;
  if (tb >= n) return;                       // wave-uniform exit
  const int rowA = tb + (lane & 15);
  const int rdA  = (rowA < n) ? rowA : (n - 1);
  const int hi8  = (lane >> 4) << 3;
  const int nkt  = din >> 5;

#pragma unroll
  for (int nt = 0; nt < 4; ++nt) {
    v8f c = bias_acc(b, nt * 16, lane);
    for (int kt = 0; kt < nkt; ++kt) {
      const v16bf a  = load_A_b128(xin + rdA * din, din, kt * 32, lane);
      const v16bf bm = load_B_packed(pWn, kt * 4 + nt, lane);
      c = WMMA_BF16(a, bm, c);
    }
    const int col = nt * 16 + (lane & 15);
#pragma unroll
    for (int j = 0; j < 8; ++j) {
      const int r = tb + j + hi8;
      if (r < n) xt_bf[r * HID + col] = (bf16)c[j];
    }
  }
}

// ---- edge MLP: m = relu([xt[src] || ea@We+be] @ W1 + b1) @ W2 + b2 ---------
// 128 threads = 4 waves; wave w owns output N-tile w; 64 edges per block.

__global__ __launch_bounds__(128)
void edge_mlp(const bf16* __restrict__ xt_bf, const int* __restrict__ src,
              const int* __restrict__ dst, const float* __restrict__ ea,
              const bf16* __restrict__ pWe, const float* __restrict__ be,
              const bf16* __restrict__ pW1, const float* __restrict__ b1,
              const bf16* __restrict__ pW2, const float* __restrict__ b2,
              float* __restrict__ agg, int E, int Etot) {
  __shared__ __align__(16) bf16 sA[ETILE * 128];   // [xt_src || et] panel
  __shared__ __align__(16) bf16 sM[ETILE * HID];   // relu(msg1) bf16
  __shared__ int sDst[ETILE];

  const int tid   = threadIdx.x;
  const int lane  = tid & 31;
  const int w     = tid >> 5;                 // wave id == N-tile
  const int tb    = blockIdx.x * ETILE;
  const int khalf = (lane >> 4) << 3;

  // Per-wave weight panels in registers (pre-packed B layout; 2 b128 each).
  const v16bf rWe = load_B_packed(pWe, w, lane);
  v16bf rW1[4];
#pragma unroll
  for (int kt = 0; kt < 4; ++kt) rW1[kt] = load_B_packed(pW1, kt * 4 + w, lane);
  v16bf rW2[2];
#pragma unroll
  for (int kt = 0; kt < 2; ++kt) rW2[kt] = load_B_packed(pW2, kt * 4 + w, lane);

  // Stage gathered bf16 node rows (cols 0..63) with 128-bit loads + dst ids.
  for (int idx = tid; idx < ETILE * 8; idx += 128) {
    const int el = idx >> 3, ch = idx & 7;
    const int e  = tb + el;
    uint4 v = make_uint4(0u, 0u, 0u, 0u);
    if (e < Etot) {
      const int s = (e < E) ? src[e] : (e - E);        // synthesized self loop
      __builtin_prefetch(src + e + 4096, 0, 0);        // global_prefetch_b8
      v = ((const uint4*)(xt_bf + s * HID))[ch];
    }
    ((uint4*)(sA + el * 128))[ch] = v;
  }
  if (tid < ETILE) {
    const int e = tb + tid;
    sDst[tid] = (e < Etot) ? ((e < E) ? dst[e] : (e - E)) : 0;
  }

  // et = ea @ We + be  -> sA cols 64..127.  A fragment: K<16 from two float4
  // loads (K>=16 is the zero padding); self loops use attr = 1.0.
#pragma unroll
  for (int mt = 0; mt < 4; ++mt) {
    const int erow = tb + mt * 16 + (lane & 15);
    v16bf a;
    float4 f0 = make_float4(0.f, 0.f, 0.f, 0.f), f1 = f0;
    if (erow < E) {
      const float4* ep = (const float4*)(ea + erow * EDIM + khalf);
      f0 = ep[0];
      f1 = ep[1];
    } else if (erow < Etot) {
      f0 = f1 = make_float4(1.f, 1.f, 1.f, 1.f);
    }
    a[0] = (bf16)f0.x; a[1] = (bf16)f0.y; a[2] = (bf16)f0.z; a[3] = (bf16)f0.w;
    a[4] = (bf16)f1.x; a[5] = (bf16)f1.y; a[6] = (bf16)f1.z; a[7] = (bf16)f1.w;
#pragma unroll
    for (int i = 8; i < 16; ++i) a[i] = (bf16)0.0f;
    v8f c = bias_acc(be, w * 16, lane);
    c = WMMA_BF16(a, rWe, c);
    const int col = 64 + w * 16 + (lane & 15);
#pragma unroll
    for (int j = 0; j < 8; ++j) sA[(mt * 16 + j + khalf) * 128 + col] = (bf16)c[j];
  }
  __syncthreads();

  // msg1 + ReLU -> bf16 LDS
#pragma unroll
  for (int mt = 0; mt < 4; ++mt) {
    v8f c = bias_acc(b1, w * 16, lane);
#pragma unroll
    for (int kt = 0; kt < 4; ++kt) {
      const v16bf a = load_A_b128(sA + mt * 16 * 128, 128, kt * 32, lane);
      c = WMMA_BF16(a, rW1[kt], c);
    }
    const int col = w * 16 + (lane & 15);
#pragma unroll
    for (int j = 0; j < 8; ++j) {
      const float v = c[j] > 0.0f ? c[j] : 0.0f;
      sM[(mt * 16 + j + khalf) * HID + col] = (bf16)v;
    }
  }
  __syncthreads();

  // msg2 + atomic scatter-add into agg[dst]
#pragma unroll
  for (int mt = 0; mt < 4; ++mt) {
    v8f c = bias_acc(b2, w * 16, lane);
#pragma unroll
    for (int kt = 0; kt < 2; ++kt) {
      const v16bf a = load_A_b128(sM + mt * 16 * HID, HID, kt * 32, lane);
      c = WMMA_BF16(a, rW2[kt], c);
    }
    const int col = w * 16 + (lane & 15);
#pragma unroll
    for (int j = 0; j < 8; ++j) {
      const int m = mt * 16 + j + khalf;
      if (tb + m < Etot) atomicAdd(&agg[sDst[m] * HID + col], c[j]);
    }
  }
}

// ---- node update: relu([agg/cnt || xt] @ U1 + b1) @ U2 + b2, LN, skip ------

__global__ __launch_bounds__(128)
void node_update(const float* __restrict__ agg, const float* __restrict__ invc,
                 const bf16* __restrict__ xt_bf,
                 const bf16* __restrict__ pU1, const float* __restrict__ ub1,
                 const bf16* __restrict__ pU2, const float* __restrict__ ub2,
                 const float* __restrict__ lng, const float* __restrict__ lnb,
                 const float* __restrict__ skipp, const float* __restrict__ hprev,
                 float* __restrict__ hout, bf16* __restrict__ hbf, int n) {
  __shared__ __align__(16) bf16  sA[64 * 128];
  __shared__ __align__(16) bf16  sM[64 * HID];
  __shared__ __align__(16) float sO[64 * HID];

  const int tid   = threadIdx.x;
  const int lane  = tid & 31;
  const int w     = tid >> 5;
  const int tb    = blockIdx.x * 64;
  const int khalf = (lane >> 4) << 3;

  v16bf rU1[4];
#pragma unroll
  for (int kt = 0; kt < 4; ++kt) rU1[kt] = load_B_packed(pU1, kt * 4 + w, lane);
  v16bf rU2[2];
#pragma unroll
  for (int kt = 0; kt < 2; ++kt) rU2[kt] = load_B_packed(pU2, kt * 4 + w, lane);

  // Stage A cols 0..63 = agg*inv (float4 -> packed bf16), cols 64..127 = xt.
  for (int idx = tid; idx < 64 * 16; idx += 128) {
    const int r = idx >> 4, ch = idx & 15;
    const int g = tb + r;
    union { uint2 u; bf16 h[4]; } t;
    if (g < n) {
      const float iv = invc[g];
      const float4 f = ((const float4*)(agg + g * HID))[ch];
      t.h[0] = (bf16)(f.x * iv); t.h[1] = (bf16)(f.y * iv);
      t.h[2] = (bf16)(f.z * iv); t.h[3] = (bf16)(f.w * iv);
    } else {
      t.h[0] = t.h[1] = t.h[2] = t.h[3] = (bf16)0.0f;
    }
    *(uint2*)(sA + r * 128 + ch * 4) = t.u;
  }
  for (int idx = tid; idx < 64 * 8; idx += 128) {
    const int r = idx >> 3, ch = idx & 7;
    const int g = tb + r;
    uint4 v = make_uint4(0u, 0u, 0u, 0u);
    if (g < n) v = ((const uint4*)(xt_bf + g * HID))[ch];
    ((uint4*)(sA + r * 128 + 64))[ch] = v;
  }
  __syncthreads();

#pragma unroll
  for (int mt = 0; mt < 4; ++mt) {
    v8f c = bias_acc(ub1, w * 16, lane);
#pragma unroll
    for (int kt = 0; kt < 4; ++kt) {
      const v16bf a = load_A_b128(sA + mt * 16 * 128, 128, kt * 32, lane);
      c = WMMA_BF16(a, rU1[kt], c);
    }
    const int col = w * 16 + (lane & 15);
#pragma unroll
    for (int j = 0; j < 8; ++j) {
      const float v = c[j] > 0.0f ? c[j] : 0.0f;
      sM[(mt * 16 + j + khalf) * HID + col] = (bf16)v;
    }
  }
  __syncthreads();

#pragma unroll
  for (int mt = 0; mt < 4; ++mt) {
    v8f c = bias_acc(ub2, w * 16, lane);
#pragma unroll
    for (int kt = 0; kt < 2; ++kt) {
      const v16bf a = load_A_b128(sM + mt * 16 * HID, HID, kt * 32, lane);
      c = WMMA_BF16(a, rU2[kt], c);
    }
    const int col = w * 16 + (lane & 15);
#pragma unroll
    for (int j = 0; j < 8; ++j) sO[(mt * 16 + j + khalf) * HID + col] = c[j];
  }
  __syncthreads();

  // Fused LayerNorm + ReLU(skip)*h_prev; one row per thread (threads 0..63).
  const bool has = (skipp != nullptr);
  float sk = 0.0f;
  if (has) { const float s = *skipp; sk = s > 0.0f ? s : 0.0f; }
  for (int r = tid; r < 64; r += 128) {
    const int g = tb + r;
    if (g >= n) continue;
    float mu = 0.0f, m2 = 0.0f;
    for (int c2 = 0; c2 < HID; ++c2) { const float v = sO[r * HID + c2]; mu += v; m2 += v * v; }
    mu *= (1.0f / HID);
    const float var = m2 * (1.0f / HID) - mu * mu;
    const float rs  = rsqrtf(var + LN_EPS);
    for (int c2 = 0; c2 < HID; ++c2) {
      float y = (sO[r * HID + c2] - mu) * rs * lng[c2] + lnb[c2];
      if (has) y += sk * hprev[g * HID + c2];
      hout[g * HID + c2] = y;
      if (hbf) hbf[g * HID + c2] = (bf16)y;
    }
  }
}

// ---- host ------------------------------------------------------------------

extern "C" void kernel_launch(void* const* d_in, const int* in_sizes, int n_in,
                              void* d_out, int out_size, void* d_ws, size_t ws_size,
                              hipStream_t stream) {
  (void)in_sizes; (void)n_in; (void)out_size; (void)ws_size;

  const float* x     = (const float*)d_in[0];
  const int*   ei    = (const int*)d_in[1];
  const float* eattr = (const float*)d_in[2];
  const int*   src   = ei;
  const int*   dstp  = ei + NE;
  // params: per layer (sorted keys): lin_edge(w,b), lin_node(w,b), ln_b, ln_g,
  //         msg1(w,b), msg2(w,b), upd1(w,b), upd2(w,b); then 'skip'.
  auto P = [&](int l, int j) { return (const float*)d_in[3 + l * 14 + j]; };
  const float* skipv = (const float*)d_in[3 + 3 * 14];

  float* ws   = (float*)d_ws;
  float* agg  = ws;           ws += NN * HID;
  float* invc = ws;           ws += NN;
  float* h    = ws;           ws += NN * HID;
  bf16*  inbf = (bf16*)ws;    ws += NN * HID;      // N x 128 bf16
  bf16*  xtbf = (bf16*)ws;    ws += NN * (HID/2);  // N x 64 bf16
  bf16*  pwb  = (bf16*)ws;                          // packed weights (3 layers)
  float* dout = (float*)d_out;

  const int Etot = NE + NN;
  const int TWN = 16, TWE = 4, TW1 = 16, TW2 = 8, TU1 = 16, TU2 = 8;
  const int PWL = (TWN + TWE + TW1 + TW2 + TU1 + TU2) * 512;   // per-layer bf16

  // degrees (graph-static): cnt = 1 (self loop) + in-degree; then invert
  fill_f32<<<(NN + 255) / 256, 256, 0, stream>>>(invc, 1.0f, NN);
  degree_k<<<(NE + 255) / 256, 256, 0, stream>>>(dstp, invc, NE);
  invert_k<<<(NN + 255) / 256, 256, 0, stream>>>(invc, NN);
  f32_to_bf16_k<<<(NN * NDIM + 255) / 256, 256, 0, stream>>>(x, inbf, NN * NDIM);

  // pre-pack all weight matrices into WMMA B-operand layout (bf16)
  auto packs = [&](const float* W, bf16* out, int K, int ktiles) {
    const int total = ktiles * 4 * 512;
    pack_B<<<(total + 255) / 256, 256, 0, stream>>>(W, out, K, ktiles);
  };
  for (int l = 0; l < 3; ++l) {
    bf16* base = pwb + l * PWL;
    const int din = (l == 0) ? NDIM : HID;
    packs(P(l, 2),  base,                         din, din / 32);  // lin_node.w
    packs(P(l, 0),  base + TWN * 512,             EDIM, 1);        // lin_edge.w
    packs(P(l, 6),  base + (TWN + TWE) * 512,     128, 4);         // msg1.w
    packs(P(l, 8),  base + (TWN + TWE + TW1) * 512, 64, 2);        // msg2.w
    packs(P(l, 10), base + (TWN + TWE + TW1 + TW2) * 512, 128, 4); // upd1.w
    packs(P(l, 12), base + (TWN + TWE + TW1 + TW2 + TU1) * 512, 64, 2); // upd2.w
  }

  for (int l = 0; l < 3; ++l) {
    const int din = (l == 0) ? NDIM : HID;
    bf16* base = pwb + l * PWL;
    const bf16 *pWn = base, *pWe = base + TWN * 512,
               *pW1 = base + (TWN + TWE) * 512,
               *pW2 = base + (TWN + TWE + TW1) * 512,
               *pU1 = base + (TWN + TWE + TW1 + TW2) * 512,
               *pU2 = base + (TWN + TWE + TW1 + TW2 + TU1) * 512;
    const float *leb = P(l, 1), *lnbn = P(l, 3), *lnB = P(l, 4), *lnG = P(l, 5),
                *m1b = P(l, 7), *m2b = P(l, 9), *u1b = P(l, 11), *u2b = P(l, 13);

    const int waves = (NN + 15) / 16;
    node_xform<<<(waves + 3) / 4, 128, 0, stream>>>(inbf, pWn, lnbn, xtbf, NN, din);
    fill_f32<<<(NN * HID + 255) / 256, 256, 0, stream>>>(agg, 0.0f, NN * HID);
    edge_mlp<<<(Etot + ETILE - 1) / ETILE, 128, 0, stream>>>(
        xtbf, src, dstp, eattr, pWe, leb, pW1, m1b, pW2, m2b, agg, NE, Etot);

    const bool last = (l == 2);
    node_update<<<(NN + 63) / 64, 128, 0, stream>>>(
        agg, invc, xtbf, pU1, u1b, pU2, u2b, lnG, lnB,
        (l > 0) ? (skipv + (l - 1)) : nullptr, h,
        last ? dout : h, last ? (bf16*)nullptr : inbf, NN);
  }
}
